// SelfAttentionV4_37314675867766
// MI455X (gfx1250) — matrix operand order repaired and checked by reading, not compiled, not run on hardware
//
#include <hip/hip_runtime.h>
#include <hip/hip_bf16.h>

// Problem constants (B=4, S=2048, D=1024, fp32 in/out)
#define S_LEN 2048
#define D_DIM 1024
#define B_SZ  4

typedef __attribute__((ext_vector_type(16))) _Float16 v16h;
typedef __attribute__((ext_vector_type(8)))  _Float16 v8h;
typedef __attribute__((ext_vector_type(8)))  float    v8f;

union AF { v16h v; v8h h[2]; };

static __device__ __forceinline__ v8f wmma_f16(v16h a, v16h b, v8f c) {
  // D = A(16x32 f16) x B(32x16 f16) + C(16x16 f32)
  return __builtin_amdgcn_wmma_f32_16x16x32_f16(
      /*neg_a=*/false, a, /*neg_b=*/false, b,
      /*c_mod=*/(short)0, c, /*reuse_a=*/false, /*reuse_b=*/false);
}

static __device__ __forceinline__ v8f vzero8() {
  v8f z;
#pragma unroll
  for (int i = 0; i < 8; ++i) z[i] = 0.0f;
  return z;
}

// ---------------------------------------------------------------------------
// f32 -> f16 conversion (weights). n must be a multiple of 256.
// ---------------------------------------------------------------------------
__global__ void __launch_bounds__(256) cvt_f32_f16(const float* __restrict__ in,
                                                   _Float16* __restrict__ out) {
  int i = blockIdx.x * 256 + threadIdx.x;
  out[i] = (_Float16)in[i];
}

// ---------------------------------------------------------------------------
// QKV projection: Out[m,n] = (f16) ( sum_k X[m,k] * W[n,k] + bias[n] )
// X fp32 (converted on the fly), W pre-converted to f16.
// Block = 256 threads (8 waves); tile = 64 rows x 128 cols; wave tile 16x64.
// A fragment (16x32, MxK): lane m = lane&15; elem i -> k = (i&7)+16*(i>=8)+8*hi
// B fragment (32x16, KxN): lane n = lane&15; elem i -> k = (i&7)+16*(i>=8)+8*hi
// C/D (16x16): elem r -> m = r + 8*hi, n = lane&15
// ---------------------------------------------------------------------------
__global__ void __launch_bounds__(256) qkv_gemm(const float* __restrict__ X,
                                                const _Float16* __restrict__ Wh,
                                                const float* __restrict__ bias,
                                                _Float16* __restrict__ Out) {
  const int wave = threadIdx.x >> 5;
  const int lane = threadIdx.x & 31;
  const int ln   = lane & 15;
  const int hi8  = (lane >> 4) << 3;   // 0 or 8
  const int hi16 = lane & 16;          // 0 or 16

  const int row0 = blockIdx.y * 64 + (wave & 3) * 16;
  const int col0 = blockIdx.x * 128 + (wave >> 2) * 64;

  v8f acc[4];
#pragma unroll
  for (int ct = 0; ct < 4; ++ct) acc[ct] = vzero8();

  const float* xrow = X + (size_t)(row0 + ln) * D_DIM;

  for (int k0 = 0; k0 < D_DIM; k0 += 32) {
    AF a;
#pragma unroll
    for (int j = 0; j < 8; ++j) {
      a.v[j]     = (_Float16)xrow[k0 + hi8 + j];        // k = j + 8*hi
      a.v[j + 8] = (_Float16)xrow[k0 + hi8 + 16 + j];   // k = 16 + j + 8*hi
    }
#pragma unroll
    for (int ct = 0; ct < 4; ++ct) {
      const _Float16* wp = Wh + (size_t)(col0 + ct * 16 + ln) * D_DIM + k0 + hi8;
      AF bm;
      bm.h[0] = *(const v8h*)wp;        // k = hi8 + 0..7
      bm.h[1] = *(const v8h*)(wp + 16); // k = hi8 + 16..23
      acc[ct] = wmma_f16(a.v, bm.v, acc[ct]);
    }
  }

#pragma unroll
  for (int ct = 0; ct < 4; ++ct) {
    const float bv = bias[col0 + ct * 16 + ln];
#pragma unroll
    for (int r = 0; r < 8; ++r) {
      const int m = row0 + r + hi8;
      Out[(size_t)m * D_DIM + col0 + ct * 16 + ln] = (_Float16)(acc[ct][r] + bv);
    }
  }
}

// ---------------------------------------------------------------------------
// Transpose V: Vh[b][s][d] -> Vt[b][d][s]  (f16, 32x32 LDS tiles)
// ---------------------------------------------------------------------------
__global__ void __launch_bounds__(256) transpose_v(const _Float16* __restrict__ Vh,
                                                   _Float16* __restrict__ Vt) {
  __shared__ _Float16 t[32][33];
  const int b  = blockIdx.z;
  const int s0 = blockIdx.x * 32;
  const int d0 = blockIdx.y * 32;
  const int x  = threadIdx.x;   // 0..31
  const int y0 = threadIdx.y;   // 0..7
  const _Float16* src = Vh + (size_t)b * S_LEN * D_DIM;
  _Float16* dst = Vt + (size_t)b * D_DIM * S_LEN;
#pragma unroll
  for (int y = y0; y < 32; y += 8) t[y][x] = src[(size_t)(s0 + y) * D_DIM + d0 + x];
  __syncthreads();
#pragma unroll
  for (int y = y0; y < 32; y += 8) dst[(size_t)(d0 + y) * S_LEN + s0 + x] = t[x][y];
}

// ---------------------------------------------------------------------------
// Flash attention (causal), one block per (batch, 16-query tile).
// 8 waves; per 256-key block:
//   prefetch: warm next k-block's K / Vt lines (global_prefetch_b8), hoisted
//             OUT of the hot loops so it cannot perturb unrolling
//   phase 1: wave w computes 16x32 score tile for keys [kb+32w, kb+32w+32)
//   phase 2: online softmax (wave w owns rows 2w, 2w+1), P -> LDS f16
//   phase 3: each wave owns a 128-wide d-slice; O += P(16x256) @ V-slice
// ---------------------------------------------------------------------------
__global__ void __launch_bounds__(256) flash_attn(const _Float16* __restrict__ Qh,
                                                  const _Float16* __restrict__ Kh,
                                                  const _Float16* __restrict__ Vt,
                                                  float* __restrict__ Out) {
  __shared__ float    s_sc[16][260];   // padded (+4 f32) to spread banks
  __shared__ _Float16 s_P[16][264];    // padded (+8 f16)
  __shared__ float    s_m[16], s_l[16], s_al[16];

  const int b    = blockIdx.y;
  const int q0   = blockIdx.x * 16;
  const int wave = threadIdx.x >> 5;
  const int lane = threadIdx.x & 31;
  const int ln   = lane & 15;
  const int hi   = lane >> 4;
  const int hi8  = hi << 3;
  const int hi16 = lane & 16;
  const float NEG_INF = -__builtin_inff();
  const float scale = 0.03125f;  // 1/sqrt(1024)

  if (threadIdx.x < 16) { s_m[threadIdx.x] = NEG_INF; s_l[threadIdx.x] = 0.0f; }
  __syncthreads();

  v8f o[8];
#pragma unroll
  for (int t = 0; t < 8; ++t) o[t] = vzero8();

  const _Float16* qbase = Qh + ((size_t)b * S_LEN + q0 + ln) * D_DIM;
  const _Float16* vtb   = Vt + (size_t)b * D_DIM * S_LEN;
  const int kmax = q0 + 15;

  for (int kb = 0; kb <= kmax; kb += 256) {
    // ---- hoisted prefetch of next k-block (speculative, partial coverage) ----
    if (kb + 256 <= kmax) {
      const int kbn = kb + 256;
      // K: one next-block row per thread, two 128B lines per row
      const _Float16* kpre = Kh + ((size_t)b * S_LEN + kbn + (int)threadIdx.x) * D_DIM;
      __builtin_prefetch(kpre, 0, 1);
      __builtin_prefetch(kpre + 512, 0, 1);
      // Vt: this wave's d-slice, next-block s-range (256 halfs = 4 lines; take 2)
      const _Float16* vpre = vtb + (size_t)(wave * 128 + lane) * S_LEN + kbn;
      __builtin_prefetch(vpre, 0, 1);
      __builtin_prefetch(vpre + 64, 0, 1);
    }

    // ---- phase 1: scores ----
    const int kk0 = kb + wave * 32;
    v8f sc0 = vzero8(), sc1 = vzero8();
    if (kk0 <= kmax) {  // whole 32-key slice beyond causal frontier -> skip
      for (int d0 = 0; d0 < D_DIM; d0 += 32) {
        AF a;
        const _Float16* qp = qbase + d0 + hi8;
        a.h[0] = *(const v8h*)qp;          // k = 0..7  (+8*hi)
        a.h[1] = *(const v8h*)(qp + 16);   // k = 16..23 (+8*hi)
        const _Float16* kp0 = Kh + ((size_t)b * S_LEN + kk0 + ln) * D_DIM + d0 + hi8;
        AF b0; b0.h[0] = *(const v8h*)kp0; b0.h[1] = *(const v8h*)(kp0 + 16);
        sc0 = wmma_f16(a.v, b0.v, sc0);
        const _Float16* kp1 = kp0 + (size_t)16 * D_DIM;
        AF b1; b1.h[0] = *(const v8h*)kp1; b1.h[1] = *(const v8h*)(kp1 + 16);
        sc1 = wmma_f16(a.v, b1.v, sc1);
      }
    }
#pragma unroll
    for (int r = 0; r < 8; ++r) {
      s_sc[r + hi8][wave * 32 + ln]      = sc0[r] * scale;
      s_sc[r + hi8][wave * 32 + 16 + ln] = sc1[r] * scale;
    }
    __syncthreads();

    // ---- phase 2: online softmax. wave w: lanes 0-15 -> row 2w, 16-31 -> 2w+1
    {
      const int row = wave * 2 + hi;
      const int q   = q0 + row;
      float vals[16];
      float mloc = NEG_INF;
#pragma unroll
      for (int t = 0; t < 16; ++t) {
        const int c = ln + (t << 4);
        float s = s_sc[row][c];
        if (kb + c > q) s = NEG_INF;   // causal mask
        vals[t] = s;
        mloc = fmaxf(mloc, s);
      }
#pragma unroll
      for (int off = 1; off < 16; off <<= 1) mloc = fmaxf(mloc, __shfl_xor(mloc, off));
      const float mprev = s_m[row];
      const float mnew  = fmaxf(mprev, mloc);   // finite: k=0 always unmasked
      float suml = 0.0f;
#pragma unroll
      for (int t = 0; t < 16; ++t) {
        const float p = __expf(vals[t] - mnew);
        suml += p;
        s_P[row][ln + (t << 4)] = (_Float16)p;
      }
#pragma unroll
      for (int off = 1; off < 16; off <<= 1) suml += __shfl_xor(suml, off);
      if (ln == 0) {
        const float al = __expf(mprev - mnew);
        s_al[row] = al;
        s_m[row]  = mnew;
        s_l[row]  = s_l[row] * al + suml;
      }
    }
    __syncthreads();

    // ---- phase 3: rescale O, accumulate P @ V (wave owns d-cols [128w,128w+128))
    {
      float al[8];
#pragma unroll
      for (int r = 0; r < 8; ++r) al[r] = s_al[r + hi8];
#pragma unroll
      for (int t = 0; t < 8; ++t)
#pragma unroll
        for (int r = 0; r < 8; ++r) o[t][r] *= al[r];

      for (int jt = 0; jt < 8; ++jt) {
        AF a;
        const _Float16* pp = &s_P[ln][(jt << 5) + hi8];
        a.h[0] = *(const v8h*)pp;          // j = 0..7  (+8*hi)
        a.h[1] = *(const v8h*)(pp + 16);   // j = 16..23 (+8*hi)
#pragma unroll
        for (int dt = 0; dt < 8; ++dt) {
          const _Float16* vp =
              vtb + (size_t)(wave * 128 + dt * 16 + ln) * S_LEN + kb + (jt << 5) + hi8;
          AF bm; bm.h[0] = *(const v8h*)vp; bm.h[1] = *(const v8h*)(vp + 16);
          o[dt] = wmma_f16(a.v, bm.v, o[dt]);
        }
      }
    }
    __syncthreads();
  }

  // ---- epilogue: O /= l ----
  float linv[8];
#pragma unroll
  for (int r = 0; r < 8; ++r) linv[r] = 1.0f / s_l[r + hi8];
#pragma unroll
  for (int t = 0; t < 8; ++t)
#pragma unroll
    for (int r = 0; r < 8; ++r)
      Out[((size_t)b * S_LEN + q0 + r + hi8) * D_DIM + wave * 128 + t * 16 + ln] =
          o[t][r] * linv[r];
}

// ---------------------------------------------------------------------------
// Launch. Inputs: X, attention_mask(unused; causal applied analytically),
// Wq, bq, Wk, bk, Wv, bv. Workspace layout (f16):
//   [Wqh | Wkh | Wvh | Qh | Kh | Vh | Vt]  = (3*1M + 4*8M)*2B ~= 73.4 MB
// ---------------------------------------------------------------------------
extern "C" void kernel_launch(void* const* d_in, const int* in_sizes, int n_in,
                              void* d_out, int out_size, void* d_ws, size_t ws_size,
                              hipStream_t stream) {
  const float* X  = (const float*)d_in[0];
  const float* Wq = (const float*)d_in[2];
  const float* bq = (const float*)d_in[3];
  const float* Wk = (const float*)d_in[4];
  const float* bk = (const float*)d_in[5];
  const float* Wv = (const float*)d_in[6];
  const float* bv = (const float*)d_in[7];
  float* Out = (float*)d_out;

  const size_t DD = (size_t)D_DIM * D_DIM;          // 1M elems
  const size_t MD = (size_t)B_SZ * S_LEN * D_DIM;   // 8M elems
  _Float16* Wqh = (_Float16*)d_ws;
  _Float16* Wkh = Wqh + DD;
  _Float16* Wvh = Wkh + DD;
  _Float16* Qh  = Wvh + DD;
  _Float16* Kh  = Qh + MD;
  _Float16* Vh  = Kh + MD;
  _Float16* Vt  = Vh + MD;

  // 1) weights f32 -> f16
  cvt_f32_f16<<<dim3(DD / 256), dim3(256), 0, stream>>>(Wq, Wqh);
  cvt_f32_f16<<<dim3(DD / 256), dim3(256), 0, stream>>>(Wk, Wkh);
  cvt_f32_f16<<<dim3(DD / 256), dim3(256), 0, stream>>>(Wv, Wvh);

  // 2) QKV projections (M=8192, N=K=1024), tile 64x128
  dim3 gg(D_DIM / 128, (B_SZ * S_LEN) / 64);
  qkv_gemm<<<gg, dim3(256), 0, stream>>>(X, Wqh, bq, Qh);
  qkv_gemm<<<gg, dim3(256), 0, stream>>>(X, Wkh, bk, Kh);
  qkv_gemm<<<gg, dim3(256), 0, stream>>>(X, Wvh, bv, Vh);

  // 3) V transpose for contiguous PV B-fragments
  transpose_v<<<dim3(S_LEN / 32, D_DIM / 32, B_SZ), dim3(32, 8), 0, stream>>>(Vh, Vt);

  // 4) causal flash attention
  flash_attn<<<dim3(S_LEN / 16, B_SZ), dim3(256), 0, stream>>>(Qh, Kh, Vt, Out);
}
